// FacialGNN_88768384073895
// MI455X (gfx1250) — matrix operand order repaired and checked by reading, not compiled, not run on hardware
//
#include <hip/hip_runtime.h>
#include <hip/hip_bf16.h>

typedef __attribute__((ext_vector_type(16))) _Float16 v16h;
typedef __attribute__((ext_vector_type(8)))  float    v8f;
typedef __attribute__((ext_vector_type(4)))  unsigned int u32x4;
typedef __attribute__((ext_vector_type(8)))  int      i32x8;
typedef __attribute__((ext_vector_type(4)))  int      i32x4;

// ---- problem constants (match reference) ----
constexpr int Bn  = 128;            // graphs
constexpr int Nn  = 68;             // nodes per graph
constexpr int Cc  = 3;              // input channels
constexpr int Pp  = 32;             // patch size
constexpr int Kk  = 16;             // conv kernels
constexpr int FDd = 64;             // feature dim
constexpr int GHh = 128;            // gcn hidden
constexpr int Tt  = Bn * Nn;        // 8704 total nodes
constexpr int Etot = Bn * 8 * Nn;   // 69632 edges
constexpr int F1  = Kk * (Pp/2) * (Pp/2);  // 4096 flattened features

// ---- workspace layout (bytes) ----
constexpr size_t OFF_FLAT  = 0;
constexpr size_t SZ_FLAT   = (size_t)Tt * F1 * 2;      // f16 pooled features
constexpr size_t OFF_FEATS = OFF_FLAT + SZ_FLAT;
constexpr size_t SZ_FEATS  = (size_t)Tt * FDd * 4;     // f32 node feats
constexpr size_t OFF_XW    = OFF_FEATS + SZ_FEATS;
constexpr size_t SZ_XW     = (size_t)Tt * GHh * 4;     // f32 nodes @ gcn_w
constexpr size_t OFF_H     = OFF_XW + SZ_XW;
constexpr size_t SZ_H      = (size_t)Tt * GHh * 4;     // f32 aggregated
constexpr size_t OFF_DEG   = OFF_H + SZ_H;
constexpr size_t SZ_DEG    = (size_t)Tt * 4;           // f32 deg -> dis
constexpr size_t OFF_G     = OFF_DEG + SZ_DEG;         // f32 pooled graphs [B,GH]

// =====================================================================
// Tensor Data Mover: 2D tile load (global -> LDS), D# per ISA 08 §8.3/8.4.
//  d0flags packs group1 dword0: data_size[17:16], pad_enable[20],
//  pad_interval[24:22], pad_amount[31:25]; workgroup_mask = 0.
// =====================================================================
__device__ __forceinline__ void tdm_load_2d(
    unsigned lds_off, const void* gaddr, unsigned d0flags,
    unsigned tdim0, unsigned tdim1, unsigned tile0, unsigned tile1,
    unsigned long long stride0)
{
    unsigned long long ga = (unsigned long long)gaddr;
    u32x4 g0;
    g0[0] = 1u;                                          // count=1, valid, user D#
    g0[1] = lds_off;                                     // LDS byte address
    g0[2] = (unsigned)(ga & 0xffffffffu);                // global_addr[31:0]
    g0[3] = (unsigned)((ga >> 32) & 0x01ffffffu)         // global_addr[56:32]
            | 0x80000000u;                               // type=2 ("image")
    i32x8 g1;
    g1[0] = (int)d0flags;
    g1[1] = (int)(tdim0 << 16);                          // tensor_dim0[15:0]
    g1[2] = (int)((tdim0 >> 16) | (tdim1 << 16));        // dim0 hi | dim1 lo
    g1[3] = (int)((tdim1 >> 16) | (tile0 << 16));        // dim1 hi | tile_dim0
    g1[4] = (int)(tile1 & 0xffffu);                      // tile_dim1 | tile_dim2=0
    g1[5] = (int)(stride0 & 0xffffffffu);                // tensor_dim0_stride lo
    g1[6] = (int)((stride0 >> 32) & 0xffffu);            // stride hi | dim1_stride=0
    g1[7] = 0;
    i32x4 z = {0, 0, 0, 0};
#if __has_include(<hip/amd_detail/amd_gfx1250_TDM.h>)
    i32x8 z8 = {0, 0, 0, 0, 0, 0, 0, 0};
    __builtin_amdgcn_tensor_load_to_lds(g0, g1, z, z, z8, 0);
#else
    __builtin_amdgcn_tensor_load_to_lds(g0, g1, z, z, 0);
#endif
}

// =====================================================================
// Kernel 1: per-node conv3x3 (SAME) + bias + relu + maxpool2x2 -> f16 flat
// Implicit GEMM: M=16 pixels, N=16 kernels, K=27 (C*3*3) padded to 32.
// One block (4 waves) per node image; 64 pixel tiles, 16 per wave.
// =====================================================================
__global__ __launch_bounds__(128) void fg_conv_pool(
    const float* __restrict__ x, const float* __restrict__ conv_w,
    const float* __restrict__ conv_b, _Float16* __restrict__ flat)
{
    const int t   = blockIdx.x;       // node id [0, T)
    const int n   = t % Nn;           // node type
    const int tid = threadIdx.x;
    const int lane = tid & 31;
    const int wave = tid >> 5;
    const int hs   = lane >> 4;       // lane half
    const int ncol = lane & 15;       // N index (kernel) for B/C/D frags

    __shared__ float    img[Cc][Pp + 2][Pp + 2];       // padded image
    __shared__ _Float16 conv_out[Kk][Pp * Pp];         // relu'd conv result

    // zero halo + load image
    for (int i = tid; i < Cc * (Pp + 2) * (Pp + 2); i += 128)
        ((float*)img)[i] = 0.0f;
    __syncthreads();
    const float* xin = x + (size_t)t * (Cc * Pp * Pp);
    for (int i = tid; i < Cc * Pp * Pp; i += 128) {
        int c = i / (Pp * Pp), rem = i % (Pp * Pp);
        img[c][rem / Pp + 1][rem % Pp + 1] = xin[i];
    }

    // B fragment: weights [K=27 pad 32][N=16], f32 -> f16
    v16h bfrag;
#pragma unroll
    for (int j = 0; j < 16; ++j) {
        int k = 16 * hs + j;
        float w = 0.0f;
        if (k < 27) {
            int c = k / 9, r = (k % 9) / 3, s = k % 3;
            w = conv_w[(((size_t)n * Kk + ncol) * Cc + c) * 9 + r * 3 + s];
        }
        bfrag[j] = (_Float16)w;
    }
    const float bias = conv_b[n * Kk + ncol];
    __syncthreads();

    // 64 M-tiles of 16 pixels, 16 per wave
    for (int ti = 0; ti < 16; ++ti) {
        const int tile = wave * 16 + ti;
        const int m  = lane & 15;
        const int p  = tile * 16 + m;
        const int py = p >> 5, px = p & 31;
        v16h afrag;
#pragma unroll
        for (int j = 0; j < 16; ++j) {
            int k = 16 * hs + j;
            float v = 0.0f;
            if (k < 27) {
                int c = k / 9, dy = (k % 9) / 3, dx = k % 3;
                v = img[c][py + dy][px + dx];
            }
            afrag[j] = (_Float16)v;
        }
        v8f acc;
#pragma unroll
        for (int r = 0; r < 8; ++r) acc[r] = bias;
        acc = __builtin_amdgcn_wmma_f32_16x16x32_f16(
            false, afrag, false, bfrag, (short)0, acc, false, false);
#pragma unroll
        for (int r = 0; r < 8; ++r) {
            int pp = tile * 16 + r + 8 * hs;
            float v = acc[r];
            conv_out[ncol][pp] = (_Float16)(v > 0.0f ? v : 0.0f);
        }
    }
    __syncthreads();

    // maxpool 2x2 + flatten: f = kk*256 + py*16 + px
    _Float16* out = flat + (size_t)t * F1;
    for (int f = tid; f < F1; f += 128) {
        int kk = f >> 8, rem = f & 255;
        int base = ((rem >> 4) * 2) * Pp + (rem & 15) * 2;
        float v0 = (float)conv_out[kk][base],      v1 = (float)conv_out[kk][base + 1];
        float v2 = (float)conv_out[kk][base + Pp], v3 = (float)conv_out[kk][base + Pp + 1];
        out[f] = (_Float16)fmaxf(fmaxf(v0, v1), fmaxf(v2, v3));
    }
}

// =====================================================================
// Kernel 2: lin1 per node type n: [128,4096] @ [4096,64] + bias, relu.
// TDM-staged K-chunks of 64: wave 0 issues tensor_load_to_lds for the
// f16 activation tile (with TDM row padding: +1 DWORD per 32-DWORD row
// to de-conflict LDS banks) and the f32 weight tile, waits TENSORcnt,
// then the block barrier publishes. 8 waves x 4 WMMA tiles.
// Dynamic LDS only -> D# lds_addr is the literal byte offset.
//   [0,           16384)  : Wstage f32 [64][64]
//   [16384,       33280)  : Astage f16, 128 rows, stride 66 halves (132 B)
// =====================================================================
constexpr unsigned LIN1_LDS_W = 0;
constexpr unsigned LIN1_LDS_A = 16384;
constexpr int      AS_STRIDE  = 66;      // halves per padded row
constexpr unsigned LIN1_LDS_BYTES = 16384 + 128 * AS_STRIDE * 2;

__global__ __launch_bounds__(256) void fg_lin1(
    const _Float16* __restrict__ flat, const float* __restrict__ lin1_w,
    const float* __restrict__ lin1_b, float* __restrict__ feats)
{
    extern __shared__ char smem[];
    float*    Ws = (float*)(smem + LIN1_LDS_W);
    _Float16* As = (_Float16*)(smem + LIN1_LDS_A);

    const int n    = blockIdx.x;
    const int tid  = threadIdx.x;
    const int lane = tid & 31;
    const int wave = tid >> 5;
    const int hs   = lane >> 4;

    v8f acc[4];
#pragma unroll
    for (int q = 0; q < 4; ++q)
#pragma unroll
        for (int r = 0; r < 8; ++r) acc[q][r] = 0.0f;

    for (int kc = 0; kc < F1; kc += 64) {
        if (kc + 64 < F1)
            __builtin_prefetch(&lin1_w[((size_t)n * F1 + kc + 64) * FDd], 0, 1);
        if (wave == 0) {
            // A tile: f16, tile 64 x 128 rows, row stride Nn*F1 halves,
            // data_size=1 (2B), pad_enable, pad_interval=32 DWORDs (code 4),
            // pad_amount=1 DWORD (code 0)  -> LDS row stride 132 B.
            tdm_load_2d(LIN1_LDS_A,
                        &flat[(size_t)n * F1 + kc],
                        (1u << 16) | (1u << 20) | (4u << 22),
                        /*tdim0=*/64, /*tdim1=*/128, /*tile0=*/64, /*tile1=*/128,
                        /*stride0=*/(unsigned long long)Nn * F1);
            // W tile: f32, tile 64 x 64, row stride 64, data_size=2 (4B).
            tdm_load_2d(LIN1_LDS_W,
                        &lin1_w[((size_t)n * F1 + kc) * FDd],
                        (2u << 16),
                        /*tdim0=*/64, /*tdim1=*/64, /*tile0=*/64, /*tile1=*/64,
                        /*stride0=*/64ull);
            __builtin_amdgcn_s_wait_tensorcnt(0);
        }
        __syncthreads();
#pragma unroll
        for (int q = 0; q < 4; ++q) {
            const int tileId = wave * 4 + q;
            const int Mt = tileId >> 2, Nt = tileId & 3;
            const int m = lane & 15;
#pragma unroll
            for (int ks = 0; ks < 64; ks += 32) {
                v16h a, bm;
#pragma unroll
                for (int j = 0; j < 16; ++j) {
                    int k = 16 * hs + j;
                    a[j] = As[(Mt * 16 + m) * AS_STRIDE + ks + k];
                }
#pragma unroll
                for (int j = 0; j < 16; ++j)
                    bm[j] = (_Float16)Ws[(ks + 16 * hs + j) * 64 + Nt * 16 + m];
                acc[q] = __builtin_amdgcn_wmma_f32_16x16x32_f16(
                    false, a, false, bm, (short)0, acc[q], false, false);
            }
        }
        __syncthreads();
    }
#pragma unroll
    for (int q = 0; q < 4; ++q) {
        const int tileId = wave * 4 + q;
        const int Mt = tileId >> 2, Nt = tileId & 3;
        const int ncol = Nt * 16 + (lane & 15);
        const float bias = lin1_b[n * FDd + ncol];
#pragma unroll
        for (int r = 0; r < 8; ++r) {
            int b = Mt * 16 + r + 8 * hs;
            float v = acc[q][r] + bias;
            feats[((size_t)b * Nn + n) * FDd + ncol] = v > 0.0f ? v : 0.0f;
        }
    }
}

// =====================================================================
// Kernel 3: xw = feats @ gcn_w : [8704,64] @ [64,128]. 128 rows / block.
// Each wave owns one N-tile (Nt=wave), sweeps 8 M-tiles, K=64 (2 steps).
// =====================================================================
__global__ __launch_bounds__(256) void fg_gcn_xw(
    const float* __restrict__ feats, const float* __restrict__ gcn_w,
    float* __restrict__ xw)
{
    const int row0 = blockIdx.x * 128;
    const int tid  = threadIdx.x;
    const int lane = tid & 31;
    const int wave = tid >> 5;       // = Nt
    const int hs   = lane >> 4;

    __shared__ _Float16 As[128][64 + 4];
    __shared__ _Float16 Ws[64][128 + 4];

    for (int i = tid; i < 128 * 64; i += 256) {
        int rr = i >> 6, k = i & 63;
        As[rr][k] = (_Float16)feats[(size_t)(row0 + rr) * FDd + k];
    }
    for (int i = tid; i < 64 * 128; i += 256) {
        int k = i >> 7, c = i & 127;
        Ws[k][c] = (_Float16)gcn_w[(size_t)k * GHh + c];
    }
    __syncthreads();

    v8f acc[8];
#pragma unroll
    for (int q = 0; q < 8; ++q)
#pragma unroll
        for (int r = 0; r < 8; ++r) acc[q][r] = 0.0f;

#pragma unroll
    for (int ks = 0; ks < 64; ks += 32) {
        v16h bm;
#pragma unroll
        for (int j = 0; j < 16; ++j)
            bm[j] = Ws[ks + 16 * hs + j][wave * 16 + (lane & 15)];
#pragma unroll
        for (int q = 0; q < 8; ++q) {
            v16h a;
#pragma unroll
            for (int j = 0; j < 16; ++j) {
                int k = 16 * hs + j;
                a[j] = As[q * 16 + (lane & 15)][ks + k];
            }
            acc[q] = __builtin_amdgcn_wmma_f32_16x16x32_f16(
                false, a, false, bm, (short)0, acc[q], false, false);
        }
    }
    const int col = wave * 16 + (lane & 15);
#pragma unroll
    for (int q = 0; q < 8; ++q)
#pragma unroll
        for (int r = 0; r < 8; ++r)
            xw[(size_t)(row0 + q * 16 + r + 8 * hs) * GHh + col] = acc[q][r];
}

// ============== graph normalization / aggregation ====================
__global__ void fg_deg_init(float* __restrict__ deg) {
    int i = blockIdx.x * blockDim.x + threadIdx.x;
    if (i < Tt) deg[i] = 1.0f;                     // self loop
}
__global__ void fg_deg_count(const int* __restrict__ dst, float* __restrict__ deg) {
    int e = blockIdx.x * blockDim.x + threadIdx.x;
    if (e < Etot)
        __hip_atomic_fetch_add(&deg[dst[e]], 1.0f,
                               __ATOMIC_RELAXED, __HIP_MEMORY_SCOPE_AGENT);
}
__global__ void fg_dis(float* __restrict__ deg) {
    int i = blockIdx.x * blockDim.x + threadIdx.x;
    if (i < Tt) deg[i] = rsqrtf(deg[i]);           // deg >= 1 always
}
__global__ __launch_bounds__(128) void fg_h_init(
    const float* __restrict__ xw, const float* __restrict__ dis,
    const float* __restrict__ gcn_b, float* __restrict__ h)
{
    int t = blockIdx.x, g = threadIdx.x;
    float d = dis[t];
    h[(size_t)t * GHh + g] = gcn_b[g] + d * d * xw[(size_t)t * GHh + g];
}
__global__ __launch_bounds__(128) void fg_edge_agg(
    const int* __restrict__ src, const int* __restrict__ dst,
    const float* __restrict__ dis, const float* __restrict__ xw,
    float* __restrict__ h)
{
    int e = blockIdx.x, g = threadIdx.x;
    int s = src[e], d = dst[e];
    float norm = dis[s] * dis[d];
    __hip_atomic_fetch_add(&h[(size_t)d * GHh + g], norm * xw[(size_t)s * GHh + g],
                           __ATOMIC_RELAXED, __HIP_MEMORY_SCOPE_AGENT);
}
__global__ __launch_bounds__(128) void fg_mean_pool(
    const float* __restrict__ h, float* __restrict__ g)
{
    int b = blockIdx.x, j = threadIdx.x;
    float s = 0.0f;
    for (int i = 0; i < Nn; ++i) {
        float v = h[(size_t)(b * Nn + i) * GHh + j];
        s += v > 0.0f ? v : 0.0f;                  // relu then mean
    }
    g[(size_t)b * GHh + j] = s / (float)Nn;
}

// =====================================================================
// Kernel: MLP head. layer1 [128,128]@[128,64] via WMMA, layer2 scalar.
// Single block, 8 waves.
// =====================================================================
__global__ __launch_bounds__(256) void fg_mlp(
    const float* __restrict__ g, const float* __restrict__ w1,
    const float* __restrict__ b1, const float* __restrict__ w2,
    const float* __restrict__ b2, float* __restrict__ out)
{
    const int tid  = threadIdx.x;
    const int lane = tid & 31;
    const int wave = tid >> 5;
    const int hs   = lane >> 4;

    __shared__ _Float16 Gs[128][128 + 4];
    __shared__ _Float16 W1s[128][64 + 4];
    __shared__ _Float16 Z[128][64];

    for (int i = tid; i < 128 * 128; i += 256) {
        int b = i >> 7, k = i & 127;
        Gs[b][k] = (_Float16)g[i];
    }
    for (int i = tid; i < 128 * 64; i += 256) {
        int k = i >> 6, c = i & 63;
        W1s[k][c] = (_Float16)w1[i];
    }
    __syncthreads();

    v8f acc[4];
#pragma unroll
    for (int q = 0; q < 4; ++q)
#pragma unroll
        for (int r = 0; r < 8; ++r) acc[q][r] = 0.0f;

#pragma unroll
    for (int q = 0; q < 4; ++q) {
        const int tileId = wave * 4 + q;
        const int Mt = tileId >> 2, Nt = tileId & 3;
        const int m = lane & 15;
#pragma unroll
        for (int ks = 0; ks < 128; ks += 32) {
            v16h a, bm;
#pragma unroll
            for (int j = 0; j < 16; ++j) {
                int k = 16 * hs + j;
                a[j] = Gs[Mt * 16 + m][ks + k];
            }
#pragma unroll
            for (int j = 0; j < 16; ++j)
                bm[j] = W1s[ks + 16 * hs + j][Nt * 16 + m];
            acc[q] = __builtin_amdgcn_wmma_f32_16x16x32_f16(
                false, a, false, bm, (short)0, acc[q], false, false);
        }
    }
#pragma unroll
    for (int q = 0; q < 4; ++q) {
        const int tileId = wave * 4 + q;
        const int Mt = tileId >> 2, Nt = tileId & 3;
        const int ncol = Nt * 16 + (lane & 15);
        const float bias = b1[ncol];
#pragma unroll
        for (int r = 0; r < 8; ++r) {
            float v = acc[q][r] + bias;
            Z[Mt * 16 + r + 8 * hs][ncol] = (_Float16)(v > 0.0f ? v : 0.0f);
        }
    }
    __syncthreads();

    // layer 2: [128,64] @ [64,2] + b2 -> out[128,2]
    int b = tid >> 1, o = tid & 1;
    float s = b2[o];
    for (int j = 0; j < 64; ++j)
        s += (float)Z[b][j] * w2[j * 2 + o];
    out[b * 2 + o] = s;
}

extern "C" void kernel_launch(void* const* d_in, const int* in_sizes, int n_in,
                              void* d_out, int out_size, void* d_ws, size_t ws_size,
                              hipStream_t stream) {
    (void)in_sizes; (void)n_in; (void)out_size; (void)ws_size;
    const float* x       = (const float*)d_in[0];
    const int*   eidx    = (const int*)d_in[1];      // [2, Etot]
    const float* conv_w  = (const float*)d_in[3];
    const float* conv_b  = (const float*)d_in[4];
    const float* lin1_w  = (const float*)d_in[5];
    const float* lin1_b  = (const float*)d_in[6];
    const float* gcn_w   = (const float*)d_in[7];
    const float* gcn_b   = (const float*)d_in[8];
    const float* mlp_w1  = (const float*)d_in[9];
    const float* mlp_b1  = (const float*)d_in[10];
    const float* mlp_w2  = (const float*)d_in[11];
    const float* mlp_b2  = (const float*)d_in[12];
    float* out = (float*)d_out;

    char* ws = (char*)d_ws;
    _Float16* flat  = (_Float16*)(ws + OFF_FLAT);
    float*    feats = (float*)(ws + OFF_FEATS);
    float*    xw    = (float*)(ws + OFF_XW);
    float*    h     = (float*)(ws + OFF_H);
    float*    dis   = (float*)(ws + OFF_DEG);
    float*    gbuf  = (float*)(ws + OFF_G);
    const int* src = eidx;
    const int* dst = eidx + Etot;

    fg_conv_pool<<<Tt, 128, 0, stream>>>(x, conv_w, conv_b, flat);
    fg_lin1<<<Nn, 256, LIN1_LDS_BYTES, stream>>>(flat, lin1_w, lin1_b, feats);
    fg_gcn_xw<<<Tt / 128, 256, 0, stream>>>(feats, gcn_w, xw);
    fg_deg_init<<<(Tt + 255) / 256, 256, 0, stream>>>(dis);
    fg_deg_count<<<(Etot + 255) / 256, 256, 0, stream>>>(dst, dis);
    fg_dis<<<(Tt + 255) / 256, 256, 0, stream>>>(dis);
    fg_h_init<<<Tt, 128, 0, stream>>>(xw, dis, gcn_b, h);
    fg_edge_agg<<<Etot, 128, 0, stream>>>(src, dst, dis, xw, h);
    fg_mean_pool<<<Bn, 128, 0, stream>>>(h, gbuf);
    fg_mlp<<<1, 256, 0, stream>>>(gbuf, mlp_w1, mlp_b1, mlp_w2, mlp_b2, out);
}
